// MultiHeadAttention_652835029612
// MI455X (gfx1250) — matrix operand order repaired and checked by reading, not compiled, not run on hardware
//
#include <hip/hip_runtime.h>

typedef __attribute__((ext_vector_type(16))) __bf16 v16bf;
typedef __attribute__((ext_vector_type(8)))  float  v8f;

typedef __attribute__((ext_vector_type(4))) unsigned int tdm_u4;
typedef __attribute__((ext_vector_type(8))) int          tdm_i8;
typedef __attribute__((ext_vector_type(4))) int          tdm_i4;

static constexpr int S_LEN   = 2048;
static constexpr int D_MODEL = 1024;
static constexpr int NHEAD   = 16;
static constexpr int D_K     = 64;
static constexpr int BATCH   = 2;

#define LDK   40          // GEMM LDS row pitch in ushort units (32 data + 8 pad)
#define LDKW  (LDK / 2)   // pitch in u32 units
#define LDKT  72          // TDM-padded pitch: 64 data + 8 pad halves (144 B)
#define LDKTW (LDKT / 2)

// round-to-nearest-even f32 -> bf16 bit pattern
__device__ __forceinline__ unsigned short f2bf(float x) {
  union { float f; unsigned u; } c; c.f = x;
  unsigned r = c.u + 0x7FFFu + ((c.u >> 16) & 1u);
  return (unsigned short)(r >> 16);
}

// Load a 16x32 bf16 A/B fragment from an LDS tile (row-major [row][k]).
// Wave32 layout (ISA 7.12.2): lane = 16*g + r ; VGPR j holds K pair at
// k = (j<4?0:16) + g*8 + (j&3)*2.
__device__ __forceinline__ v16bf frag_ld(const unsigned short* smem, int row0,
                                         int lane, int pitch_u32) {
  const unsigned* rp = (const unsigned*)smem
                     + (row0 + (lane & 15)) * pitch_u32
                     + ((lane >> 4) & 1) * 4;
  union { v16bf v; unsigned u[8]; } f;
#pragma unroll
  for (int j = 0; j < 8; ++j)
    f.u[j] = rp[((j & 4) << 1) + (j & 3)];
  return f.v;
}

__device__ __forceinline__ v8f wmma_bf16(v16bf a, v16bf b, v8f c) {
  return __builtin_amdgcn_wmma_f32_16x16x32_bf16(false, a, false, b, (short)0, c, false, false);
}

// ---------------------------------------------------------------------------
// TDM: async-load a 2D bf16 tile (tile_d1 rows x tile_d0 elems, row stride
// stride0 elems) from global into LDS with hardware padding:
// every 32 DWORDs (128B row) insert 4 DWORDs (16B) -> LDS pitch 144B = LDKT.
// Descriptor per CDNA5 ISA ch.8. Uniform args; tracked by TENSORcnt.
// This toolchain exposes the 6-arg builtin (g0, g1, g2, g3, g4, cpol).
// ---------------------------------------------------------------------------
__device__ __forceinline__ void tdm_load_tile_bf16(
    unsigned lds_addr, const void* gptr,
    unsigned tensor_d0, unsigned tensor_d1,
    unsigned tile_d0, unsigned tile_d1, unsigned stride0) {
  unsigned long long ga = (unsigned long long)(size_t)gptr;
  tdm_u4 g0;
  g0[0] = 1u;                                      // count=1 (valid), user mode
  g0[1] = lds_addr;                                // LDS byte address
  g0[2] = (unsigned)(ga & 0xffffffffu);            // global_addr[31:0]
  g0[3] = (unsigned)((ga >> 32) & 0x01ffffffu)     // global_addr[56:32]
        | (2u << 30);                              // type = 2 ("image")
  tdm_i8 g1;
  g1[0] = (int)((1u << 16)                         // data_size = 2 bytes
              | (1u << 20)                         // pad_enable
              | (4u << 22)                         // pad_interval: 32 DWORDs
              | (3u << 25));                       // pad_amount: 4 DWORDs
  g1[1] = (int)((tensor_d0 & 0xffffu) << 16);      // tensor_dim0[15:0]
  g1[2] = (int)(((tensor_d0 >> 16) & 0xffffu)      // tensor_dim0[31:16]
              | ((tensor_d1 & 0xffffu) << 16));    // tensor_dim1[15:0]
  g1[3] = (int)(((tensor_d1 >> 16) & 0xffffu)      // tensor_dim1[31:16]
              | ((tile_d0 & 0xffffu) << 16));      // tile_dim0
  g1[4] = (int)(tile_d1 & 0xffffu);                // tile_dim1 (tile_dim2 = 0)
  g1[5] = (int)stride0;                            // tensor_dim0_stride[31:0]
  g1[6] = 0;                                       // stride0 hi, dim1_stride lo
  g1[7] = 0;
  tdm_i4 gz4 = {0, 0, 0, 0};
  tdm_i8 gz8 = {0, 0, 0, 0, 0, 0, 0, 0};
  __builtin_amdgcn_tensor_load_to_lds(g0, g1, gz4, gz4, gz8, 0);
}

// ---------------------------------------------------------------------------
// Generic Y = A * W^T (+bias) GEMM, M=4096, N=K=1024, 64x64 tile, 4 waves,
// each wave owns a 32x32 subtile (2x2 WMMA fragments).
// ---------------------------------------------------------------------------
template <bool A_BF16, bool OUT_SPLIT>
__global__ __launch_bounds__(128) void gemm_wt_kernel(
    const void* __restrict__ Ap, const float* __restrict__ W,
    const float* __restrict__ bias, void* __restrict__ Yp) {
  constexpr int K = D_MODEL;
  constexpr int N = D_MODEL;
  __shared__ __align__(16) unsigned short As[64 * LDK];
  __shared__ __align__(16) unsigned short Bs[64 * LDK];
  const int tid = threadIdx.x, lane = tid & 31, wave = tid >> 5;
  const int m0 = blockIdx.y * 64, n0 = blockIdx.x * 64;
  const int wm = (wave >> 1) * 32, wn = (wave & 1) * 32;
  v8f c00 = {}, c01 = {}, c10 = {}, c11 = {};

  for (int k0 = 0; k0 < K; k0 += 32) {
    __syncthreads();
    if (A_BF16) {
      const unsigned short* A = (const unsigned short*)Ap;
      for (int i = tid; i < 64 * 4; i += 128) {
        int r = i >> 2, c8 = (i & 3) * 8;
        *(uint4*)(As + r * LDK + c8) =
            *(const uint4*)(A + (size_t)(m0 + r) * K + k0 + c8);
        if (k0 + 32 < K)
          __builtin_prefetch(A + (size_t)(m0 + r) * K + k0 + 32 + c8, 0, 1);
      }
    } else {
      const float* A = (const float*)Ap;
      for (int i = tid; i < 64 * 8; i += 128) {
        int r = i >> 3, c4 = (i & 7) * 4;
        float4 x = *(const float4*)(A + (size_t)(m0 + r) * K + k0 + c4);
        unsigned short* d = As + r * LDK + c4;
        d[0] = f2bf(x.x); d[1] = f2bf(x.y); d[2] = f2bf(x.z); d[3] = f2bf(x.w);
        if (k0 + 32 < K)
          __builtin_prefetch(A + (size_t)(m0 + r) * K + k0 + 32 + c4, 0, 1);
      }
    }
    // B fragment wants B[k][n] = W[n][k] -> stage W rows: Bs[n][k] = W[n0+n][k0+k]
    for (int i = tid; i < 64 * 8; i += 128) {
      int r = i >> 3, c4 = (i & 7) * 4;
      float4 x = *(const float4*)(W + (size_t)(n0 + r) * K + k0 + c4);
      unsigned short* d = Bs + r * LDK + c4;
      d[0] = f2bf(x.x); d[1] = f2bf(x.y); d[2] = f2bf(x.z); d[3] = f2bf(x.w);
      if (k0 + 32 < K)
        __builtin_prefetch(W + (size_t)(n0 + r) * K + k0 + 32 + c4, 0, 1);
    }
    __syncthreads();
    v16bf a0 = frag_ld(As, wm, lane, LDKW), a1 = frag_ld(As, wm + 16, lane, LDKW);
    v16bf b0 = frag_ld(Bs, wn, lane, LDKW), b1 = frag_ld(Bs, wn + 16, lane, LDKW);
    c00 = wmma_bf16(a0, b0, c00);
    c01 = wmma_bf16(a0, b1, c01);
    c10 = wmma_bf16(a1, b0, c10);
    c11 = wmma_bf16(a1, b1, c11);
  }

  const int mr = ((lane >> 4) & 1) * 8, nc = lane & 15;
  const v8f* cs[4] = { &c00, &c01, &c10, &c11 };
#pragma unroll
  for (int qf = 0; qf < 4; ++qf) {
    int mb = m0 + wm + (qf >> 1) * 16 + mr;
    int nb = n0 + wn + (qf & 1) * 16 + nc;
    float bv = bias[nb];
#pragma unroll
    for (int j = 0; j < 8; ++j) {
      int m = mb + j;
      float val = (*cs[qf])[j] + bv;
      if (OUT_SPLIT) {
        int b = m >> 11, s = m & (S_LEN - 1);
        int h = nb >> 6, dk = nb & (D_K - 1);
        ((unsigned short*)Yp)[(((size_t)(b * NHEAD + h) * S_LEN) + s) * D_K + dk] = f2bf(val);
      } else {
        ((float*)Yp)[(size_t)m * N + nb] = val;
      }
    }
  }
}

// ---------------------------------------------------------------------------
// scores[b,h,m,n] = (qh[b,h,m,:] . kh[b,h,n,:]) / 8, masked to -inf.
// Tiles staged by the Tensor Data Mover (async DMA with LDS padding);
// wave 0 issues two descriptors, waits on TENSORcnt, barrier releases all.
// ---------------------------------------------------------------------------
__global__ __launch_bounds__(128) void scores_kernel(
    const unsigned short* __restrict__ qh, const unsigned short* __restrict__ kh,
    const int* __restrict__ mask, float* __restrict__ attn) {
  __shared__ __align__(16) unsigned short As[64 * LDKT];
  __shared__ __align__(16) unsigned short Bs[64 * LDKT];
  const int tid = threadIdx.x, lane = tid & 31, wave = tid >> 5;
  const int bh = blockIdx.z, b = bh >> 4;
  const unsigned short* A    = qh + (size_t)bh * S_LEN * D_K;
  const unsigned short* Bsrc = kh + (size_t)bh * S_LEN * D_K;
  float* out = attn + (size_t)bh * S_LEN * S_LEN;
  const int m0 = blockIdx.y * 64, n0 = blockIdx.x * 64;
  const int wm = (wave >> 1) * 32, wn = (wave & 1) * 32;
  v8f c00 = {}, c01 = {}, c10 = {}, c11 = {};

  if (wave == 0) {
    // 64 rows x 64 bf16 tiles of qh and kh, hw-padded to LDKT-half pitch
    tdm_load_tile_bf16((unsigned)(size_t)(void*)As, A + (size_t)m0 * D_K,
                       D_K, S_LEN, D_K, 64, D_K);
    tdm_load_tile_bf16((unsigned)(size_t)(void*)Bs, Bsrc + (size_t)n0 * D_K,
                       D_K, S_LEN, D_K, 64, D_K);
    __builtin_amdgcn_s_wait_tensorcnt(0);
  }
  __syncthreads();

#pragma unroll
  for (int k0 = 0; k0 < D_K; k0 += 32) {
    v16bf a0 = frag_ld(As + k0, wm, lane, LDKTW);
    v16bf a1 = frag_ld(As + k0, wm + 16, lane, LDKTW);
    v16bf b0 = frag_ld(Bs + k0, wn, lane, LDKTW);
    v16bf b1 = frag_ld(Bs + k0, wn + 16, lane, LDKTW);
    c00 = wmma_bf16(a0, b0, c00);
    c01 = wmma_bf16(a0, b1, c01);
    c10 = wmma_bf16(a1, b0, c10);
    c11 = wmma_bf16(a1, b1, c11);
  }

  const float scale = 0.125f;  // 1/sqrt(64)
  const int mr = ((lane >> 4) & 1) * 8, nc = lane & 15;
  const v8f* cs[4] = { &c00, &c01, &c10, &c11 };
#pragma unroll
  for (int qf = 0; qf < 4; ++qf) {
    int mb = m0 + wm + (qf >> 1) * 16 + mr;
    int n  = n0 + wn + (qf & 1) * 16 + nc;
    int mv = mask[b * S_LEN + n];
#pragma unroll
    for (int j = 0; j < 8; ++j) {
      float val = mv ? (*cs[qf])[j] * scale : -__builtin_inff();
      out[(size_t)(mb + j) * S_LEN + n] = val;
    }
  }
}

// ---------------------------------------------------------------------------
// In-place row softmax over 2048-wide rows; one 256-thread block per row.
// ---------------------------------------------------------------------------
__global__ __launch_bounds__(256) void softmax_kernel(float* __restrict__ attn) {
  __shared__ float red[256];
  float* p = attn + (size_t)blockIdx.x * S_LEN;
  const int t = threadIdx.x;
  float v[8];
  float mx = -__builtin_inff();
#pragma unroll
  for (int j = 0; j < 8; ++j) { v[j] = p[t + j * 256]; mx = fmaxf(mx, v[j]); }
  red[t] = mx; __syncthreads();
  for (int s = 128; s > 0; s >>= 1) {
    if (t < s) red[t] = fmaxf(red[t], red[t + s]);
    __syncthreads();
  }
  mx = red[0]; __syncthreads();
  float sum = 0.f;
#pragma unroll
  for (int j = 0; j < 8; ++j) { v[j] = __expf(v[j] - mx); sum += v[j]; }
  red[t] = sum; __syncthreads();
  for (int s = 128; s > 0; s >>= 1) {
    if (t < s) red[t] += red[t + s];
    __syncthreads();
  }
  float inv = 1.0f / red[0];
#pragma unroll
  for (int j = 0; j < 8; ++j) p[t + j * 256] = v[j] * inv;
}

// ---------------------------------------------------------------------------
// ctx[b,s,h,:] = attn[b,h,s,:] @ vh[b,h,:,:]   (M=2048, N=64, K=2048 per b,h)
// ---------------------------------------------------------------------------
__global__ __launch_bounds__(128) void av_kernel(
    const float* __restrict__ attn, const unsigned short* __restrict__ vh,
    unsigned short* __restrict__ ctx) {
  __shared__ __align__(16) unsigned short As[64 * LDK];
  __shared__ __align__(16) unsigned short Bs[64 * LDK];
  const int tid = threadIdx.x, lane = tid & 31, wave = tid >> 5;
  const int bh = blockIdx.y, b = bh >> 4, h = bh & 15;
  const float* A = attn + (size_t)bh * S_LEN * S_LEN;
  const unsigned short* V = vh + (size_t)bh * S_LEN * D_K;
  const int m0 = blockIdx.x * 64;
  const int wm = (wave >> 1) * 32, wn = (wave & 1) * 32;  // N=64 total
  v8f c00 = {}, c01 = {}, c10 = {}, c11 = {};

  for (int k0 = 0; k0 < S_LEN; k0 += 32) {
    __syncthreads();
    for (int i = tid; i < 64 * 8; i += 128) {
      int r = i >> 3, c4 = (i & 7) * 4;
      float4 x = *(const float4*)(A + (size_t)(m0 + r) * S_LEN + k0 + c4);
      unsigned short* d = As + r * LDK + c4;
      d[0] = f2bf(x.x); d[1] = f2bf(x.y); d[2] = f2bf(x.z); d[3] = f2bf(x.w);
      if (k0 + 32 < S_LEN)
        __builtin_prefetch(A + (size_t)(m0 + r) * S_LEN + k0 + 32 + c4, 0, 1);
    }
    // Bs[n][k] = vh[k0+k][n]  (transpose stage; coalesced global reads over n)
    for (int i = tid; i < 32 * 64; i += 128) {
      int kk = i >> 6, nn = i & 63;
      Bs[nn * LDK + kk] = V[(size_t)(k0 + kk) * D_K + nn];
    }
    __syncthreads();
    v16bf a0 = frag_ld(As, wm, lane, LDKW), a1 = frag_ld(As, wm + 16, lane, LDKW);
    v16bf b0 = frag_ld(Bs, wn, lane, LDKW), b1 = frag_ld(Bs, wn + 16, lane, LDKW);
    c00 = wmma_bf16(a0, b0, c00);
    c01 = wmma_bf16(a0, b1, c01);
    c10 = wmma_bf16(a1, b0, c10);
    c11 = wmma_bf16(a1, b1, c11);
  }

  const int mr = ((lane >> 4) & 1) * 8, nc = lane & 15;
  const v8f* cs[4] = { &c00, &c01, &c10, &c11 };
#pragma unroll
  for (int qf = 0; qf < 4; ++qf) {
    int mb = m0 + wm + (qf >> 1) * 16 + mr;
    int n  = wn + (qf & 1) * 16 + nc;
#pragma unroll
    for (int j = 0; j < 8; ++j) {
      int m = mb + j;  // query index within (b,h)
      ctx[(((size_t)b * S_LEN + m) * NHEAD + h) * D_K + n] = f2bf((*cs[qf])[j]);
    }
  }
}

// ---------------------------------------------------------------------------
extern "C" void kernel_launch(void* const* d_in, const int* in_sizes, int n_in,
                              void* d_out, int out_size, void* d_ws, size_t ws_size,
                              hipStream_t stream) {
  const float* q   = (const float*)d_in[0];
  const float* k   = (const float*)d_in[1];
  const float* v   = (const float*)d_in[2];
  const int*  mask = (const int*)d_in[3];
  const float* w_q = (const float*)d_in[4];
  const float* b_q = (const float*)d_in[5];
  const float* w_v = (const float*)d_in[6];
  const float* b_v = (const float*)d_in[7];
  const float* w_o = (const float*)d_in[8];
  const float* b_o = (const float*)d_in[9];

  float* out  = (float*)d_out;                                   // [B,S,1024]
  float* attn = out + (size_t)BATCH * S_LEN * D_MODEL;           // [B,H,S,S]

  const size_t headElems = (size_t)BATCH * NHEAD * S_LEN * D_K;  // 4,194,304
  unsigned short* qh  = (unsigned short*)d_ws;
  unsigned short* kh  = qh + headElems;
  unsigned short* vh  = kh + headElems;
  unsigned short* ctx = vh + headElems;

  dim3 blk(128);
  dim3 gproj(D_MODEL / 64, (BATCH * S_LEN) / 64);

  gemm_wt_kernel<false, true ><<<gproj, blk, 0, stream>>>(q, w_q, b_q, qh);
  gemm_wt_kernel<false, true ><<<gproj, blk, 0, stream>>>(k, w_q, b_q, kh);
  gemm_wt_kernel<false, true ><<<gproj, blk, 0, stream>>>(v, w_v, b_v, vh);

  scores_kernel<<<dim3(S_LEN / 64, S_LEN / 64, BATCH * NHEAD), blk, 0, stream>>>(qh, kh, mask, attn);

  softmax_kernel<<<dim3(BATCH * NHEAD * S_LEN), dim3(256), 0, stream>>>(attn);

  av_kernel<<<dim3(S_LEN / 64, BATCH * NHEAD), blk, 0, stream>>>(attn, vh, ctx);

  gemm_wt_kernel<true, false><<<gproj, blk, 0, stream>>>(ctx, w_o, b_o, out);
}
